// SingleLayerSwin_69990787056014
// MI455X (gfx1250) — matrix operand order repaired and checked by reading, not compiled, not run on hardware
//
#include <hip/hip_runtime.h>
#include <hip/hip_bf16.h>

typedef __attribute__((ext_vector_type(16))) _Float16 v16h;
typedef __attribute__((ext_vector_type(8)))  _Float16 v8h;
typedef __attribute__((ext_vector_type(8)))  float    v8f;
typedef __attribute__((ext_vector_type(4)))  int      v4i;

#define DEV_INLINE __device__ __forceinline__

// ---------------------------------------------------------------------------
// Async global->LDS staging (CDNA5 GLOBAL_LOAD_ASYNC_TO_LDS_B128, ASYNCcnt)
// Builtin signature (from hipcc diagnostic): (v4i* gsrc, v4i* lds, i32, i32)
// Guarded: falls back to VGPR round-trip copies if the builtin is absent.
// ---------------------------------------------------------------------------
#if __has_builtin(__builtin_amdgcn_global_load_async_to_lds_b128)
#define HAVE_ASYNC_LDS 1
DEV_INLINE void async_cp16(_Float16* lds_dst, const _Float16* gsrc) {
  __builtin_amdgcn_global_load_async_to_lds_b128((v4i*)gsrc, (v4i*)lds_dst, 0, 0);
}
DEV_INLINE void async_wait0() {
  asm volatile("s_wait_asynccnt 0" ::: "memory");
}
#else
#define HAVE_ASYNC_LDS 0
#endif

// ---------------------------------------------------------------------------
// WMMA helpers (gfx1250: v_wmma_f32_16x16x32_f16, wave32)
// ---------------------------------------------------------------------------
DEV_INLINE v8f wmma16(v16h a, v16h b, v8f c) {
  // (neg_a, A, neg_b, B, c_mod, C, reuse_a, reuse_b)
  return __builtin_amdgcn_wmma_f32_16x16x32_f16(false, a, false, b, (short)0, c,
                                                false, false);
}

// A fragment: 16(M) x 32(K) tile from a row-major [rows][ldk] f16 array.
// ISA layout: lane holds row (lane&15); lanes 0-15 cover K={0..7,16..23},
// lanes 16-31 cover K={8..15,24..31} -> two contiguous 8-half (16B) loads.
DEV_INLINE v16h frag_a(const _Float16* base, int ldk, int row0, int k0, int lane) {
  int m  = row0 + (lane & 15);
  int kb = k0 + ((lane >> 4) << 3);
  const _Float16* p = base + m * ldk + kb;
  v8h lo = *(const v8h*)p;
  v8h hi = *(const v8h*)(p + 16);
  return __builtin_shufflevector(lo, hi, 0, 1, 2, 3, 4, 5, 6, 7,
                                 8, 9, 10, 11, 12, 13, 14, 15);
}

// B fragment: B_wmma[k][n] = S[n][k] where S is row-major [N][ldk] f16
// (i.e. GEMM computes A * S^T).  ISA layout: lane holds column n=(lane&15),
// lanes 0-15 cover K={0..7,16..23}, lanes 16-31 cover K={8..15,24..31}.
DEV_INLINE v16h frag_b(const _Float16* base, int ldk, int n0, int k0, int lane) {
  int n = n0 + (lane & 15);
  const _Float16* p = base + n * ldk + k0 + ((lane >> 4) << 3);
  v8h lo = *(const v8h*)p;
  v8h hi = *(const v8h*)(p + 16);
  return __builtin_shufflevector(lo, hi, 0, 1, 2, 3, 4, 5, 6, 7,
                                 8, 9, 10, 11, 12, 13, 14, 15);
}

// ---------------------------------------------------------------------------
// f32 -> f16 weight conversion
// ---------------------------------------------------------------------------
__global__ void cvt_f16(const float* __restrict__ s, _Float16* __restrict__ d,
                        int n) {
  int i = blockIdx.x * 256 + threadIdx.x;
  if (i < n) d[i] = (_Float16)s[i];
}

// ---------------------------------------------------------------------------
// LayerNorm (+ optional shift-roll + window partition), f32 -> f16
// one wave (32 lanes) per token, lane covers channels c, c+32, c+64 (C=96)
// ---------------------------------------------------------------------------
__global__ void __launch_bounds__(256)
ln_kernel(const float* __restrict__ x, const float* __restrict__ g,
          const float* __restrict__ bb, _Float16* __restrict__ out,
          int shift, int do_part) {
  const int wave = threadIdx.x >> 5, lane = threadIdx.x & 31;
  const long long tok = (long long)blockIdx.x * 8 + wave;
  const float* xp = x + tok * 96;
  float v0 = xp[lane], v1 = xp[lane + 32], v2 = xp[lane + 64];
  float s = v0 + v1 + v2;
#pragma unroll
  for (int m = 16; m; m >>= 1) s += __shfl_xor(s, m, 32);
  float mean = s * (1.0f / 96.0f);
  float d0 = v0 - mean, d1 = v1 - mean, d2 = v2 - mean;
  float q = d0 * d0 + d1 * d1 + d2 * d2;
#pragma unroll
  for (int m = 16; m; m >>= 1) q += __shfl_xor(q, m, 32);
  float rstd = rsqrtf(q * (1.0f / 96.0f) + 1e-5f);
  long long dtok = tok;
  if (do_part) {
    int bimg = (int)(tok >> 16);
    int rem  = (int)(tok & 65535);
    int hh = ((rem >> 8) - shift) & 255;   // roll by -shift
    int ww = ((rem & 255) - shift) & 255;
    dtok = (((long long)bimg << 8) + ((hh >> 4) << 4) + (ww >> 4)) * 256 +
           ((hh & 15) << 4) + (ww & 15);
  }
  _Float16* op = out + dtok * 96;
  op[lane]      = (_Float16)(d0 * rstd * g[lane]      + bb[lane]);
  op[lane + 32] = (_Float16)(d1 * rstd * g[lane + 32] + bb[lane + 32]);
  op[lane + 64] = (_Float16)(d2 * rstd * g[lane + 64] + bb[lane + 64]);
}

// ---------------------------------------------------------------------------
// Generic WMMA GEMM:  out[M,Nc] = A[M,K] * W[Nc,K]^T + bias, fused epilogues
// block = 256 threads (8 waves), 128-row block tile, A-block + W in LDS
// ---------------------------------------------------------------------------
#define GEMM_QKV  0
#define GEMM_FC1  1
#define GEMM_PROJ 2
#define GEMM_FC2  3

__global__ void __launch_bounds__(256)
gemm_wmma(const _Float16* __restrict__ A, const _Float16* __restrict__ W,
          const float* __restrict__ bias, int Nc, int K, int mode, int shift,
          _Float16* __restrict__ out_h, const float* __restrict__ resid,
          float* __restrict__ out_f) {
  extern __shared__ _Float16 sm[];
  _Float16* As = sm;            // 128*K
  _Float16* Ws = sm + 128 * K;  // Nc*K
  const int tid = threadIdx.x;
  {
    const _Float16* gA = A + (long long)blockIdx.x * 128 * K;
    const int cntA = (128 * K) >> 3;
    const int cntW = (Nc * K) >> 3;
#if HAVE_ASYNC_LDS
    for (int i = tid; i < cntA; i += 256) async_cp16(As + i * 8, gA + i * 8);
    for (int i = tid; i < cntW; i += 256) async_cp16(Ws + i * 8, W + i * 8);
    async_wait0();
#else
    const v8h* vA = (const v8h*)gA;
    v8h* sA = (v8h*)As;
    for (int i = tid; i < cntA; i += 256) sA[i] = vA[i];
    const v8h* vW = (const v8h*)W;
    v8h* sW = (v8h*)Ws;
    for (int i = tid; i < cntW; i += 256) sW[i] = vW[i];
#endif
  }
  __syncthreads();

  const int wave = tid >> 5, lane = tid & 31, half = lane >> 4;
  const int mloc = wave * 16;
  const long long gRow0 = (long long)blockIdx.x * 128 + mloc + half * 8;
  const int nTiles = Nc >> 4;

  for (int nt = 0; nt < nTiles; ++nt) {
    v8f acc = {};
    for (int k0 = 0; k0 < K; k0 += 32) {
      v16h a = frag_a(As, K, mloc, k0, lane);
      v16h b = frag_b(Ws, K, nt * 16, k0, lane);
      acc = wmma16(a, b, acc);
    }
    const int col = nt * 16 + (lane & 15);
    const float bv = bias[col];
#pragma unroll
    for (int r = 0; r < 8; ++r) {
      const long long m = gRow0 + r;       // C layout: row = r + 8*(lane>>4)
      float v = acc[r] + bv;
      if (mode == GEMM_QKV) {
        if (col < 96) v *= 0.17677669529663687f;  // SCALE = 32^-0.5 on q
        out_h[m * Nc + col] = (_Float16)v;
      } else if (mode == GEMM_FC1) {
        v = 0.5f * v * (1.0f + erff(v * 0.70710678118654752f));  // exact GELU
        out_h[m * Nc + col] = (_Float16)v;
      } else if (mode == GEMM_PROJ) {
        // window reverse + un-shift + residual add -> f32
        int wi = (int)(m >> 8), n = (int)(m & 255);
        int b_ = wi >> 8, wrem = wi & 255;
        int hh = ((wrem >> 4) << 4) + (n >> 4);
        int ww = ((wrem & 15) << 4) + (n & 15);
        int h  = (hh + shift) & 255;
        int w_ = (ww + shift) & 255;
        long long dst = (((long long)b_ << 16) + (h << 8) + w_) * 96 + col;
        out_f[dst] = resid[dst] + v;
      } else {  // GEMM_FC2: residual add, identity mapping
        long long dst = m * 96 + col;
        out_f[dst] = resid[dst] + v;
      }
    }
  }
}

// ---------------------------------------------------------------------------
// Windowed attention: one block per window (1024), 8 waves, loops 3 heads.
// S = q*k^T (one WMMA per 16x16 tile, K=HD=32), rel-pos bias + shifted mask,
// softmax via half-wave shuffles, P (f16, LDS) @ V via WMMA.
// ---------------------------------------------------------------------------
DEV_INLINE int sw_region(int p) { return p < 240 ? 0 : (p < 248 ? 1 : 2); }

__global__ void __launch_bounds__(256)
attn_kernel(const _Float16* __restrict__ qkv, const float* __restrict__ rpb,
            _Float16* __restrict__ out, int shifted) {
  extern __shared__ _Float16 sm[];
  _Float16* qs  = sm;           // 256 x 32
  _Float16* ks  = sm + 8192;    // 256 x 32
  _Float16* vTs = sm + 16384;   // 32 x 256 (transposed V)
  _Float16* Ps  = sm + 24576;   // 8 waves x 16 x 256
  const int tid  = threadIdx.x;
  const int wi   = blockIdx.x;
  const int hwnd = (wi & 255) >> 4, wwnd = wi & 15;
  const int wave = tid >> 5, lane = tid & 31, half = lane >> 4;
  const int mloc = wave * 16;

  for (int head = 0; head < 3; ++head) {
    {  // cooperative load: thread t = token t of this window
      const _Float16* row = qkv + ((long long)wi * 256 + tid) * 288 + head * 32;
#if HAVE_ASYNC_LDS
#pragma unroll
      for (int i = 0; i < 4; ++i) {
        async_cp16(qs + tid * 32 + i * 8, row + i * 8);        // q
        async_cp16(ks + tid * 32 + i * 8, row + 96 + i * 8);   // k
      }
#else
      const v8h* rq = (const v8h*)row;
      const v8h* rk = (const v8h*)(row + 96);
      v8h* dq = (v8h*)(qs + tid * 32);
      v8h* dk = (v8h*)(ks + tid * 32);
#pragma unroll
      for (int i = 0; i < 4; ++i) { dq[i] = rq[i]; dk[i] = rk[i]; }
#endif
      const v8h* rv = (const v8h*)(row + 192);
      v8h tv[4];
#pragma unroll
      for (int i = 0; i < 4; ++i) tv[i] = rv[i];
      const _Float16* te = (const _Float16*)tv;
#pragma unroll
      for (int d = 0; d < 32; ++d) vTs[d * 256 + tid] = te[d];  // transpose V
#if HAVE_ASYNC_LDS
      async_wait0();
#endif
    }
    __syncthreads();

    // ---- scores: wave owns query rows [mloc, mloc+16) ----
    float sreg[16][8];
    v16h aq = frag_a(qs, 32, mloc, 0, lane);
#pragma unroll
    for (int j = 0; j < 16; ++j) {
      v8f acc = {};
      v16h bk = frag_b(ks, 32, j * 16, 0, lane);  // B = k^T
      acc = wmma16(aq, bk, acc);
      const int nk = j * 16 + (lane & 15);
      const int r2 = nk >> 4, c2 = nk & 15;
      const int regk =
          sw_region(hwnd * 16 + r2) * 3 + sw_region(wwnd * 16 + c2);
#pragma unroll
      for (int r = 0; r < 8; ++r) {
        const int nq = mloc + r + half * 8;
        const int r1 = nq >> 4, c1 = nq & 15;
        float s =
            acc[r] + rpb[((r1 - r2 + 15) * 31 + (c1 - c2 + 15)) * 3 + head];
        if (shifted) {
          int regq = sw_region(hwnd * 16 + r1) * 3 + sw_region(wwnd * 16 + c1);
          if (regq != regk) s -= 100.0f;
        }
        sreg[j][r] = s;
      }
    }
    // ---- softmax (row = 16 lanes of one half-wave) ----
    float rmax[8], rsum[8];
#pragma unroll
    for (int r = 0; r < 8; ++r) {
      float mx = sreg[0][r];
#pragma unroll
      for (int j = 1; j < 16; ++j) mx = fmaxf(mx, sreg[j][r]);
#pragma unroll
      for (int m = 1; m < 16; m <<= 1) mx = fmaxf(mx, __shfl_xor(mx, m, 16));
      rmax[r] = mx;
    }
#pragma unroll
    for (int r = 0; r < 8; ++r) {
      float sum = 0.0f;
      const int prow = mloc + r + half * 8;
#pragma unroll
      for (int j = 0; j < 16; ++j) {
        float p = expf(sreg[j][r] - rmax[r]);
        sum += p;
        Ps[prow * 256 + j * 16 + (lane & 15)] = (_Float16)p;
      }
#pragma unroll
      for (int m = 1; m < 16; m <<= 1) sum += __shfl_xor(sum, m, 16);
      rsum[r] = 1.0f / sum;
    }
    asm volatile("" ::: "memory");  // order P LDS stores before frag reads

    // ---- O = P @ V ----
#pragma unroll
    for (int dt = 0; dt < 2; ++dt) {
      v8f acc = {};
#pragma unroll
      for (int kk = 0; kk < 8; ++kk) {
        v16h ap = frag_a(Ps + mloc * 256, 256, 0, kk * 32, lane);
        v16h bv = frag_b(vTs, 256, dt * 16, kk * 32, lane);  // B = V (via V^T)
        acc = wmma16(ap, bv, acc);
      }
      const int dcol = dt * 16 + (lane & 15);
#pragma unroll
      for (int r = 0; r < 8; ++r) {
        const int nq = mloc + r + half * 8;
        out[((long long)wi * 256 + nq) * 96 + head * 32 + dcol] =
            (_Float16)(acc[r] * rsum[r]);
      }
    }
    __syncthreads();  // before next head reuses LDS
  }
}

// ---------------------------------------------------------------------------
// Host orchestration
// ---------------------------------------------------------------------------
static constexpr size_t OFF_X  = 0;                          // f16 M*96
static constexpr size_t OFF_B  = OFF_X + 50331648;           // f16 M*384
static constexpr size_t OFF_WQ = OFF_B + 201326592;          // f16 288*96
static constexpr size_t OFF_WP = OFF_WQ + 55296;             // f16 96*96
static constexpr size_t OFF_W1 = OFF_WP + 18432;             // f16 384*96
static constexpr size_t OFF_W2 = OFF_W1 + 73728;             // f16 96*384

extern "C" void kernel_launch(void* const* d_in, const int* in_sizes, int n_in,
                              void* d_out, int out_size, void* d_ws,
                              size_t ws_size, hipStream_t stream) {
  (void)in_sizes; (void)n_in; (void)out_size; (void)ws_size;
  const float* x_in   = (const float*)d_in[0];
  const float* ln1_g  = (const float*)d_in[1];
  const float* ln1_b  = (const float*)d_in[2];
  const float* qkv_w  = (const float*)d_in[3];
  const float* qkv_b  = (const float*)d_in[4];
  const float* proj_w = (const float*)d_in[5];
  const float* proj_b = (const float*)d_in[6];
  const float* rpb    = (const float*)d_in[7];
  const float* ln2_g  = (const float*)d_in[8];
  const float* ln2_b  = (const float*)d_in[9];
  const float* fc1_w  = (const float*)d_in[10];
  const float* fc1_b  = (const float*)d_in[11];
  const float* fc2_w  = (const float*)d_in[12];
  const float* fc2_b  = (const float*)d_in[13];

  float* xcur = (float*)d_out;  // running residual buffer
  char* ws = (char*)d_ws;
  _Float16* bufX = (_Float16*)(ws + OFF_X);
  _Float16* bufB = (_Float16*)(ws + OFF_B);
  _Float16* wq = (_Float16*)(ws + OFF_WQ);
  _Float16* wp = (_Float16*)(ws + OFF_WP);
  _Float16* w1 = (_Float16*)(ws + OFF_W1);
  _Float16* w2 = (_Float16*)(ws + OFF_W2);

  for (int i = 0; i < 2; ++i) {
    const int shift = i ? 8 : 0;
    const float* xsrc = i ? (const float*)d_out : x_in;

    cvt_f16<<<(27648 + 255) / 256, 256, 0, stream>>>(qkv_w + i * 27648, wq, 27648);
    cvt_f16<<<(9216  + 255) / 256, 256, 0, stream>>>(proj_w + i * 9216,  wp, 9216);
    cvt_f16<<<(36864 + 255) / 256, 256, 0, stream>>>(fc1_w + i * 36864, w1, 36864);
    cvt_f16<<<(36864 + 255) / 256, 256, 0, stream>>>(fc2_w + i * 36864, w2, 36864);

    // LN1 + shift + window partition
    ln_kernel<<<32768, 256, 0, stream>>>(xsrc, ln1_g + i * 96, ln1_b + i * 96,
                                         bufX, shift, 1);
    // QKV GEMM (q scaled)
    gemm_wmma<<<2048, 256, (128 + 288) * 96 * 2, stream>>>(
        bufX, wq, qkv_b + i * 288, 288, 96, GEMM_QKV, 0, bufB, nullptr, nullptr);
    // windowed attention
    attn_kernel<<<1024, 256, (24576 + 32768) * 2, stream>>>(
        bufB, rpb + i * 2883, bufX, shift ? 1 : 0);
    // proj GEMM + window reverse + un-shift + residual -> xcur (f32)
    gemm_wmma<<<2048, 256, (128 + 96) * 96 * 2, stream>>>(
        bufX, wp, proj_b + i * 96, 96, 96, GEMM_PROJ, shift, nullptr, xsrc, xcur);
    // LN2 (identity layout)
    ln_kernel<<<32768, 256, 0, stream>>>(xcur, ln2_g + i * 96, ln2_b + i * 96,
                                         bufX, 0, 0);
    // fc1 + GELU
    gemm_wmma<<<2048, 256, (128 + 384) * 96 * 2, stream>>>(
        bufX, w1, fc1_b + i * 384, 384, 96, GEMM_FC1, 0, bufB, nullptr, nullptr);
    // fc2 + residual
    gemm_wmma<<<2048, 256, (128 + 96) * 384 * 2, stream>>>(
        bufB, w2, fc2_b + i * 96, 96, 384, GEMM_FC2, 0, nullptr, xcur, xcur);
  }
}
